// CrossAttentionHeightSplit_57526791963211
// MI455X (gfx1250) — matrix operand order repaired and checked by reading, not compiled, not run on hardware
//
#include <hip/hip_runtime.h>

// ---------------------------------------------------------------------------
// CDNA5 (gfx1250) cross-attention: 2 scenes x 26 views x 8 heads, C=256, S=1024
// All matrix math via v_wmma_f32_16x16x32_bf16 (wave32).
// Attention K-blocks staged with GLOBAL_LOAD_ASYNC_TO_LDS_B128 (ASYNCcnt),
// double-buffered software pipeline.
// ---------------------------------------------------------------------------

typedef __attribute__((ext_vector_type(16))) __bf16 v16bf;
typedef __attribute__((ext_vector_type(8)))  float  v8f;
typedef int v4i_g __attribute__((vector_size(16)));   // matches builtin proto

#define SEQ   1024
#define CH    256
#define NVIEW 26
#define NHEAD 8
#define DHEAD 32
#define NSLOT 66

static constexpr size_t QELEMS  = (size_t)2 * NVIEW * SEQ * CH;   // 13,631,488
static constexpr size_t KVELEMS = (size_t)NSLOT * 2 * SEQ * CH;   // 34,603,008

#if defined(__HIP_DEVICE_COMPILE__) && __has_builtin(__builtin_amdgcn_global_load_async_to_lds_b128)
#define USE_ASYNC_LDS 1
#define GLOBAL_TO_LDS_ASYNC_B128(gsrc, ldst)                                   \
  __builtin_amdgcn_global_load_async_to_lds_b128(                              \
      (__attribute__((address_space(1))) v4i_g*)(gsrc),                        \
      (__attribute__((address_space(3))) v4i_g*)(ldst), 0, 0)
#else
#define USE_ASYNC_LDS 0
#endif

#if defined(__HIP_DEVICE_COMPILE__) && __has_builtin(__builtin_amdgcn_s_wait_asynccnt)
#define ASYNC_WAIT0() __builtin_amdgcn_s_wait_asynccnt(0)
#else
#define ASYNC_WAIT0() asm volatile("s_wait_asynccnt 0x0" ::: "memory")
#endif

// ---- static attention graph (precomputed from _angle_select / _mha_index) ----
__device__ const int MHA_OF_VIEW[NVIEW] = {
  0,1, 2,2,2,2,2,2,2,2, 3,3,3,3,3,3,3,3, 4,4,4,4,4,4,4,4};

// unique (weight-set, source-view) K/V projection jobs
__device__ const int KV_SLOT_VIEW[NSLOT] = {
  18,20,22,24,                                              // m=0
  2,4,6,8,                                                  // m=1
  1,2,3,4,5,6,7,8,9,10,11,12,13,14,15,16,17,                // m=2
  2,3,4,5,6,7,8,9,10,11,12,13,14,15,16,17,18,19,20,21,22,23,24,25, // m=3
  0,10,11,12,13,14,15,16,17,18,19,20,21,22,23,24,25};       // m=4
__device__ const int KV_SLOT_M[NSLOT] = {
  0,0,0,0, 1,1,1,1,
  2,2,2,2,2,2,2,2,2,2,2,2,2,2,2,2,2,
  3,3,3,3,3,3,3,3,3,3,3,3,3,3,3,3,3,3,3,3,3,3,3,3,
  4,4,4,4,4,4,4,4,4,4,4,4,4,4,4,4,4};

__device__ const int NSEL[NVIEW] = {4,4,4,3,4,3,4,3,4,3,4,4,4,4,4,4,4,4,4,3,4,3,4,3,4,3};
__device__ const int SLOTS[NVIEW][4] = {
  { 0, 1, 2, 3},{ 4, 5, 6, 7},
  { 8,10,16,17},{ 9,11,18,-1},{ 8,10,12,19},{11,13,20,-1},
  { 8,12,14,21},{13,15,22,-1},{ 8,14,16,23},{ 9,15,24,-1},
  {25,34,40,41},{26,33,35,42},{27,34,36,43},{28,35,37,44},
  {29,36,38,45},{30,37,39,46},{31,38,40,47},{32,33,39,48},
  {49,50,59,65},{51,58,60,-1},{49,52,59,61},{53,60,62,-1},
  {49,54,61,63},{55,62,64,-1},{49,56,63,65},{57,58,64,-1}};

__device__ __forceinline__ v8f wmma_bf16(v16bf a, v16bf b, v8f c) {
  return __builtin_amdgcn_wmma_f32_16x16x32_bf16(false, a, false, b,
                                                 (short)0, c, false, false);
}

// ===========================================================================
// Kernel 1: Q/K/V projections.  158 jobs x 2 scenes; each job is a
// (1024 x 256) @ (256 x 256)^T GEMM.  A is read transposed from x ([c][s]),
// W is [cout][cin] row-major.  128x128 output tile per 256-thread block.
// ===========================================================================
__global__ __launch_bounds__(256) void fused_qkv_proj_wmma(
    const float* __restrict__ x, const float* __restrict__ w_qkv,
    const float* __restrict__ b_qkv, __bf16* __restrict__ qp,
    __bf16* __restrict__ kp, __bf16* __restrict__ vp)
{
  __shared__ __align__(16) __bf16 lds_a[128 * 32];  // [s][k]
  __shared__ __align__(16) __bf16 lds_b[128 * 32];  // [n][k]

  const int tid   = threadIdx.x;
  const int lane  = tid & 31;
  const int wave  = tid >> 5;
  const int scene = blockIdx.z & 1;
  const int job   = blockIdx.z >> 1;

  int m, view, rowoff;
  __bf16* dst;
  if (job < 26) {
    view = job; m = MHA_OF_VIEW[view]; rowoff = 0;
    dst = qp + (size_t)(scene * NVIEW + view) * SEQ * CH;
  } else if (job < 92) {
    const int slot = job - 26;
    view = KV_SLOT_VIEW[slot]; m = KV_SLOT_M[slot]; rowoff = 256;
    dst = kp + (size_t)(slot * 2 + scene) * SEQ * CH;
  } else {
    const int slot = job - 92;
    view = KV_SLOT_VIEW[slot]; m = KV_SLOT_M[slot]; rowoff = 512;
    dst = vp + (size_t)(slot * 2 + scene) * SEQ * CH;
  }
  const float* xv   = x + (size_t)(scene * NVIEW + view) * CH * SEQ;  // [c][s]
  const float* W    = w_qkv + (size_t)m * 768 * CH + (size_t)rowoff * CH;
  const float* bias = b_qkv + (size_t)m * 768 + rowoff;

  const int n0 = blockIdx.x * 128;
  const int s0 = blockIdx.y * 128;
  const int wM = (wave & 3) * 32;   // 4 M-groups of 32
  const int wN = (wave >> 2) * 64;  // 2 N-groups of 64
  const int mrow  = lane & 15;
  const int lhalf = lane >> 4;

  v8f acc[2][4];
  #pragma unroll
  for (int mt = 0; mt < 2; ++mt)
    #pragma unroll
    for (int nt = 0; nt < 4; ++nt) acc[mt][nt] = v8f{};

  for (int kk = 0; kk < CH; kk += 32) {
    // stage + fp32 -> bf16 convert
    #pragma unroll
    for (int it = 0; it < 16; ++it) {
      const int idx = it * 256 + tid;
      const int cc = idx >> 7, ss = idx & 127;        // A: 32 x 128
      lds_a[ss * 32 + cc] = (__bf16)xv[(size_t)(kk + cc) * SEQ + s0 + ss];
      const int nn = idx >> 5, kq = idx & 31;         // B: 128 x 32
      lds_b[nn * 32 + kq] = (__bf16)W[(size_t)(n0 + nn) * CH + kk + kq];
    }
    __syncthreads();

    v16bf af[2], bf[4];
    #pragma unroll
    for (int mt = 0; mt < 2; ++mt)
      #pragma unroll
      for (int i = 0; i < 16; ++i) {
        const int k = 16 * lhalf + i;  // A layout
        af[mt][i] = lds_a[(wM + mt * 16 + mrow) * 32 + k];
      }
    #pragma unroll
    for (int nt = 0; nt < 4; ++nt)
      #pragma unroll
      for (int i = 0; i < 16; ++i)                             // B layout
        bf[nt][i] = lds_b[(wN + nt * 16 + mrow) * 32 + 16 * lhalf + i];

    #pragma unroll
    for (int mt = 0; mt < 2; ++mt)
      #pragma unroll
      for (int nt = 0; nt < 4; ++nt)
        acc[mt][nt] = wmma_bf16(af[mt], bf[nt], acc[mt][nt]);
    __syncthreads();
  }

  #pragma unroll
  for (int mt = 0; mt < 2; ++mt)
    #pragma unroll
    for (int nt = 0; nt < 4; ++nt)
      #pragma unroll
      for (int r = 0; r < 8; ++r) {
        const int row = s0 + wM + mt * 16 + r + 8 * lhalf;  // D layout
        const int col = n0 + wN + nt * 16 + mrow;
        dst[(size_t)row * CH + col] = (__bf16)(acc[mt][nt][r] + bias[col]);
      }
}

// ===========================================================================
// Kernel 2: flash attention.  One block = (scene,view,head,128 q rows);
// each wave owns 16 q rows.  64-key blocks, double-buffered:
//   K block  -> LDS via GLOBAL_LOAD_ASYNC_TO_LDS_B128 (ASYNCcnt)
//   V block  -> global_load_b128 one block early, transposed commit to LDS
// ===========================================================================
__global__ __launch_bounds__(256) void flash_cross_attn_wmma(
    const __bf16* __restrict__ qp, const __bf16* __restrict__ kp,
    const __bf16* __restrict__ vp, __bf16* __restrict__ op)
{
  __shared__ __align__(16) __bf16 lds_k[2][64 * 32];    // [key][d]
  __shared__ __align__(16) __bf16 lds_v[2][32 * 64];    // [d][key] (transposed)
  __shared__ __align__(16) __bf16 lds_p[8 * 16 * 64];   // per-wave P tiles

  const int tid  = threadIdx.x;
  const int lane = tid & 31;
  const int wave = tid >> 5;
  const int z    = blockIdx.z;            // scene*26 + view
  const int view = z % NVIEW;
  const int scene = z / NVIEW;
  const int head = blockIdx.y;
  const int q0   = blockIdx.x * 128 + wave * 16;
  const int nsel = NSEL[view];
  const int mrow  = lane & 15;
  const int lhalf = lane >> 4;
  // softmax scale 1/sqrt(32) folded with log2(e) so we can use exp2
  const float SC = 0.17677669529663687f * 1.4426950408889634f;

  // staging geometry: 64 rows x 64B, one 16B chunk per thread
  const int kr = tid >> 2, pt = tid & 3;

  // Q fragment (A layout, 16x32, pre-scaled)
  v16bf qf;
  {
    const __bf16* qb = qp + ((size_t)z * SEQ + q0 + mrow) * CH + head * DHEAD;
    #pragma unroll
    for (int i = 0; i < 16; ++i) {
      const int d = 16 * lhalf + i;
      qf[i] = (__bf16)((float)qb[d] * SC);
    }
  }

  float mrun[8], lrun[8];
  v8f acc0 = v8f{}, acc1 = v8f{};
  #pragma unroll
  for (int r = 0; r < 8; ++r) { mrun[r] = -1e30f; lrun[r] = 0.f; }

  const int nblk = nsel * 16;  // 64-key blocks across all selected views

  // per-thread global offset of this thread's 16B chunk within block `blk`
  auto chunk_off = [&](int blk) -> size_t {
    const int slot = SLOTS[view][blk >> 4];
    return (size_t)(slot * 2 + scene) * SEQ * CH + (size_t)head * DHEAD
         + (size_t)((blk & 15) * 64 + kr) * CH + pt * 8;
  };

  // ---- pipeline prologue: issue block 0 ----
  uint4 vreg;
  {
    const size_t off = chunk_off(0);
#if USE_ASYNC_LDS
    GLOBAL_TO_LDS_ASYNC_B128(kp + off, &lds_k[0][kr * 32 + pt * 8]);
#else
    *(uint4*)&lds_k[0][kr * 32 + pt * 8] = *(const uint4*)(kp + off);
#endif
    vreg = *(const uint4*)(vp + off);
  }

  for (int b = 0; b < nblk; ++b) {
    const int buf = b & 1;
    // commit pending V block: transpose 8 bf16 into [d][key]
    {
      const __bf16* ve = (const __bf16*)&vreg;
      #pragma unroll
      for (int e = 0; e < 8; ++e) lds_v[buf][(pt * 8 + e) * 64 + kr] = ve[e];
    }
#if USE_ASYNC_LDS
    ASYNC_WAIT0();                       // K block b landed in LDS
#endif
    __syncthreads();                     // all waves' stages visible

    // issue block b+1 (overlaps with compute below)
    if (b + 1 < nblk) {
      const int nbuf = (b + 1) & 1;
      const size_t off = chunk_off(b + 1);
#if USE_ASYNC_LDS
      GLOBAL_TO_LDS_ASYNC_B128(kp + off, &lds_k[nbuf][kr * 32 + pt * 8]);
#else
      *(uint4*)&lds_k[nbuf][kr * 32 + pt * 8] = *(const uint4*)(kp + off);
#endif
      vreg = *(const uint4*)(vp + off);
    }

    // ===== compute on buffer `buf` =====
    // scores: 16 q rows x 64 keys (4 WMMAs, contraction over d=32)
    v8f sc[4];
    #pragma unroll
    for (int g = 0; g < 4; ++g) {
      v16bf kf;
      #pragma unroll
      for (int i = 0; i < 16; ++i)
        kf[i] = lds_k[buf][(g * 16 + mrow) * 32 + 16 * lhalf + i];
      v8f zz = v8f{};
      sc[g] = wmma_bf16(qf, kf, zz);
    }

    // online softmax; rows 0-7 live in lanes 0-15, rows 8-15 in lanes 16-31
    float p[4][8];
    #pragma unroll
    for (int r = 0; r < 8; ++r) {
      float mx = fmaxf(fmaxf(sc[0][r], sc[1][r]), fmaxf(sc[2][r], sc[3][r]));
      #pragma unroll
      for (int msk = 1; msk < 16; msk <<= 1) mx = fmaxf(mx, __shfl_xor(mx, msk, 32));
      const float mn   = fmaxf(mrun[r], mx);
      const float corr = exp2f(mrun[r] - mn);
      float rs = 0.f;
      #pragma unroll
      for (int g = 0; g < 4; ++g) {
        const float pv = exp2f(sc[g][r] - mn);
        p[g][r] = pv; rs += pv;
      }
      #pragma unroll
      for (int msk = 1; msk < 16; msk <<= 1) rs += __shfl_xor(rs, msk, 32);
      lrun[r] = lrun[r] * corr + rs;
      acc0[r] *= corr; acc1[r] *= corr;
      mrun[r] = mn;
    }

    // P: D layout -> LDS -> A layout (DS ops are in-order within a wave)
    __bf16* pw = &lds_p[wave * 16 * 64];
    #pragma unroll
    for (int g = 0; g < 4; ++g)
      #pragma unroll
      for (int r = 0; r < 8; ++r)
        pw[(r + 8 * lhalf) * 64 + g * 16 + mrow] = (__bf16)p[g][r];

    // O += P @ V   (contraction over keys: 2 chunks of 32)
    #pragma unroll
    for (int ch = 0; ch < 2; ++ch) {
      v16bf pa;
      #pragma unroll
      for (int i = 0; i < 16; ++i)
        pa[i] = pw[mrow * 64 + ch * 32 + 16 * lhalf + i];
      v16bf v0f, v1f;
      #pragma unroll
      for (int i = 0; i < 16; ++i) {
        v0f[i] = lds_v[buf][(mrow)      * 64 + ch * 32 + 16 * lhalf + i];
        v1f[i] = lds_v[buf][(16 + mrow) * 64 + ch * 32 + 16 * lhalf + i];
      }
      acc0 = wmma_bf16(pa, v0f, acc0);
      acc1 = wmma_bf16(pa, v1f, acc1);
    }
  }

  #pragma unroll
  for (int r = 0; r < 8; ++r) {
    const float inv = 1.0f / lrun[r];
    const size_t row = (size_t)z * SEQ + q0 + r + 8 * lhalf;
    op[row * CH + head * DHEAD + mrow]      = (__bf16)(acc0[r] * inv);
    op[row * CH + head * DHEAD + 16 + mrow] = (__bf16)(acc1[r] * inv);
  }
}

// ===========================================================================
// Kernel 3: output projection  Y = O @ Wo^T + bo, stored transposed into
// d_out as (bt, C, H*W) fp32.
// ===========================================================================
__global__ __launch_bounds__(256) void out_proj_wmma(
    const __bf16* __restrict__ op, const float* __restrict__ w_out,
    const float* __restrict__ b_out, float* __restrict__ out)
{
  __shared__ __align__(16) __bf16 lds_a[128 * 32];
  __shared__ __align__(16) __bf16 lds_b[128 * 32];

  const int tid  = threadIdx.x;
  const int lane = tid & 31;
  const int wave = tid >> 5;
  const int z    = blockIdx.z;        // scene*26 + view
  const int view = z % NVIEW;
  const int m    = MHA_OF_VIEW[view];

  const __bf16* A    = op + (size_t)z * SEQ * CH;        // [s][c]
  const float*  W    = w_out + (size_t)m * CH * CH;      // [n][k]
  const float*  bias = b_out + (size_t)m * CH;

  const int n0 = blockIdx.x * 128;
  const int s0 = blockIdx.y * 128;
  const int wM = (wave & 3) * 32;
  const int wN = (wave >> 2) * 64;
  const int mrow  = lane & 15;
  const int lhalf = lane >> 4;

  v8f acc[2][4];
  #pragma unroll
  for (int mt = 0; mt < 2; ++mt)
    #pragma unroll
    for (int nt = 0; nt < 4; ++nt) acc[mt][nt] = v8f{};

  for (int kk = 0; kk < CH; kk += 32) {
    #pragma unroll
    for (int it = 0; it < 16; ++it) {
      const int idx = it * 256 + tid;
      const int ss = idx >> 5, cc = idx & 31;
      lds_a[ss * 32 + cc] = A[(size_t)(s0 + ss) * CH + kk + cc];
      lds_b[ss * 32 + cc] = (__bf16)W[(size_t)(n0 + ss) * CH + kk + cc];
    }
    __syncthreads();

    v16bf af[2], bf[4];
    #pragma unroll
    for (int mt = 0; mt < 2; ++mt)
      #pragma unroll
      for (int i = 0; i < 16; ++i) {
        const int k = 16 * lhalf + i;
        af[mt][i] = lds_a[(wM + mt * 16 + mrow) * 32 + k];
      }
    #pragma unroll
    for (int nt = 0; nt < 4; ++nt)
      #pragma unroll
      for (int i = 0; i < 16; ++i)
        bf[nt][i] = lds_b[(wN + nt * 16 + mrow) * 32 + 16 * lhalf + i];

    #pragma unroll
    for (int mt = 0; mt < 2; ++mt)
      #pragma unroll
      for (int nt = 0; nt < 4; ++nt)
        acc[mt][nt] = wmma_bf16(af[mt], bf[nt], acc[mt][nt]);
    __syncthreads();
  }

  #pragma unroll
  for (int mt = 0; mt < 2; ++mt)
    #pragma unroll
    for (int nt = 0; nt < 4; ++nt)
      #pragma unroll
      for (int r = 0; r < 8; ++r) {
        const int row = s0 + wM + mt * 16 + r + 8 * lhalf;
        const int col = n0 + wN + nt * 16 + mrow;
        out[((size_t)z * CH + col) * SEQ + row] = acc[mt][nt][r] + bias[col];
      }
}

// ===========================================================================
extern "C" void kernel_launch(void* const* d_in, const int* in_sizes, int n_in,
                              void* d_out, int out_size, void* d_ws, size_t ws_size,
                              hipStream_t stream) {
  (void)in_sizes; (void)n_in; (void)out_size; (void)ws_size;
  const float* x     = (const float*)d_in[0];
  const float* w_qkv = (const float*)d_in[1];
  const float* b_qkv = (const float*)d_in[2];
  const float* w_out = (const float*)d_in[3];
  const float* b_out = (const float*)d_in[4];
  float* out = (float*)d_out;

  char* wsb = (char*)d_ws;
  __bf16* qp = (__bf16*)wsb;
  __bf16* kp = (__bf16*)(wsb + QELEMS * 2);
  __bf16* vp = (__bf16*)(wsb + (QELEMS + KVELEMS) * 2);
  __bf16* op = (__bf16*)(wsb + (QELEMS + 2 * KVELEMS) * 2);

  // 1) projections: 158 jobs x 2 scenes; tiles: 2 (N) x 8 (M)
  fused_qkv_proj_wmma<<<dim3(2, 8, 316), 256, 0, stream>>>(x, w_qkv, b_qkv, qp, kp, vp);
  // 2) flash attention: 8 q-blocks x 8 heads x 52 (scene,view)
  flash_cross_attn_wmma<<<dim3(8, 8, 52), 256, 0, stream>>>(qp, kp, vp, op);
  // 3) output projection: 2 (N) x 8 (M) x 52
  out_proj_wmma<<<dim3(2, 8, 52), 256, 0, stream>>>(op, w_out, b_out, out);
}
